// EncoderLayer_15908558864752
// MI455X (gfx1250) — compile-verified, matrix operand order checked by reading, not run on hardware
//
#include <hip/hip_runtime.h>

// ---------------------------------------------------------------------------
// Types for CDNA5 WMMA (gfx1250, wave32)
// ---------------------------------------------------------------------------
typedef __bf16 bf16;
typedef __attribute__((ext_vector_type(8)))  __bf16 v8bf;
typedef __attribute__((ext_vector_type(16))) __bf16 v16bf;
typedef __attribute__((ext_vector_type(8)))  float  v8f;

union FragAB { v16bf v; v8bf h[2]; };

#define S_LEN 4096
#define D_DIM 512
#define F_DIM 2048
#define NHEAD 8
#define DH    64

static __device__ __forceinline__ v8f wmma_bf16(const v16bf& a, const v16bf& b, const v8f& c) {
    return __builtin_amdgcn_wmma_f32_16x16x32_bf16(false, a, false, b, (short)0, c, false, false);
}

// ---------------------------------------------------------------------------
// fp32 -> bf16 conversion (RNE via compiler cast), 4 elems/thread
// ---------------------------------------------------------------------------
__global__ __launch_bounds__(256) void cvt_f32_bf16(const float* __restrict__ s,
                                                    bf16* __restrict__ d, int n) {
    int i = (blockIdx.x * 256 + threadIdx.x) * 4;
    if (i + 3 < n) {
        float4 v = *(const float4*)(s + i);
        d[i + 0] = (bf16)v.x;
        d[i + 1] = (bf16)v.y;
        d[i + 2] = (bf16)v.z;
        d[i + 3] = (bf16)v.w;
    }
}

// ---------------------------------------------------------------------------
// mask[S][S] int -> transposed additive bias biasT[t][m] = (mask[m][t]==0)
// ? -1e9 : 0.  Transposed so the attention kernel can load each lane's 8
// consecutive query-rows as one v8f straight into the WMMA C operand.
// 32x32 LDS tile transpose; block = 256 threads, each handles 4 elements.
// ---------------------------------------------------------------------------
__global__ __launch_bounds__(256) void mask_bias_transpose(
    const int* __restrict__ mask, float* __restrict__ biasT) {
    __shared__ float tile[32][33];
    const int bt = blockIdx.x * 32;            // source col (t) base
    const int bm = blockIdx.y * 32;            // source row (m) base
    const int tx = threadIdx.x & 31;
    const int ty = (threadIdx.x >> 5) * 4;     // 8 waves x 4 rows
#pragma unroll
    for (int i = 0; i < 4; ++i) {
        int m = bm + ty + i;
        tile[ty + i][tx] = (mask[(size_t)m * S_LEN + bt + tx] == 0) ? -1.0e9f : 0.0f;
    }
    __syncthreads();
#pragma unroll
    for (int i = 0; i < 4; ++i) {
        int t = bt + ty + i;
        biasT[(size_t)t * S_LEN + bm + tx] = tile[tx][ty + i];
    }
}

// ---------------------------------------------------------------------------
// bf16 WMMA GEMM: C = A[MxK] * B[KxN] + bias (+relu) (+fp32 residual)
// Block tile 128x128x32, 8 waves (4x2), wave tile 32x64 (2x4 WMMA accums).
// Double-buffered LDS with register staging: one barrier per K-step; global
// loads for tile k+1 issued before the WMMAs of tile k; L2 prefetch of k+2.
//
// Compile-time epilogue:
//   OUTB_MODE: 0=no bf16 out, 1=row-major [M][N],
//              2=head-split [(n/64)*M + m][64], 3=transpose [n*M + m]
//   RELU:      apply max(v,0) before store
//   USE_RES:   v += residual[m][n]; outF[m][n] = v  (fp32 path)
// ---------------------------------------------------------------------------
template <int OUTB_MODE, bool RELU, bool USE_RES>
__global__ __launch_bounds__(256) void gemm_bf16_kernel(
    const bf16* __restrict__ A, const bf16* __restrict__ B,
    const float* __restrict__ bias, const float* __restrict__ residual,
    float* __restrict__ outF, bf16* __restrict__ outB,
    int M, int N, int K)
{
    __shared__ __align__(16) bf16 lA[2][128][32];   // row-major A tiles (16KB)
    __shared__ __align__(16) bf16 lB[2][128][32];   // B tiles stored [n][k] (16KB)

    const int tid  = threadIdx.x;
    const int wave = tid >> 5;
    const int lane = tid & 31;
    const int wm   = wave >> 1;          // 0..3
    const int wn   = wave & 1;           // 0..1
    const int lrow = lane & 15;
    const int lhalf = lane >> 4;
    const int bm = blockIdx.y * 128;
    const int bn = blockIdx.x * 128;

    const v8f vzero = {0.f, 0.f, 0.f, 0.f, 0.f, 0.f, 0.f, 0.f};
    v8f acc[2][4];
#pragma unroll
    for (int i = 0; i < 2; ++i)
#pragma unroll
        for (int j = 0; j < 4; ++j) acc[i][j] = vzero;

    // staging addresses: each thread owns 16 bf16 of A and 16 bf16 of B per tile
    const int ar = tid >> 1, ac = (tid & 1) * 16;
    const int bk = tid >> 3, bc = (tid & 7) * 16;
    const bf16* aptr = A + (size_t)(bm + ar) * K + ac;
    const bf16* bptr = B + (size_t)bk * N + bn + bc;

    // preload tile 0 into registers
    v8bf ra0 = *(const v8bf*)(aptr);
    v8bf ra1 = *(const v8bf*)(aptr + 8);
    v8bf rb0 = *(const v8bf*)(bptr);
    v8bf rb1 = *(const v8bf*)(bptr + 8);

    int buf = 0;
    for (int k0 = 0; k0 < K; k0 += 32) {
        // --- commit staged registers to LDS[buf]
        *(v8bf*)&lA[buf][ar][ac]     = ra0;
        *(v8bf*)&lA[buf][ar][ac + 8] = ra1;
#pragma unroll
        for (int i = 0; i < 8; ++i) {
            lB[buf][bc + i][bk]     = rb0[i];
            lB[buf][bc + 8 + i][bk] = rb1[i];
        }
        // --- issue global loads for next tile (overlap with WMMA below)
        if (k0 + 32 < K) {
            ra0 = *(const v8bf*)(aptr + k0 + 32);
            ra1 = *(const v8bf*)(aptr + k0 + 40);
            rb0 = *(const v8bf*)(bptr + (size_t)(k0 + 32) * N);
            rb1 = *(const v8bf*)(bptr + (size_t)(k0 + 32) * N + 8);
            if (k0 + 64 < K) {          // warm L2 two tiles ahead
                __builtin_prefetch(aptr + k0 + 64, 0, 0);
                __builtin_prefetch(bptr + (size_t)(k0 + 64) * N, 0, 0);
            }
        }
        __syncthreads();

        FragAB af[2], bfr[4];
#pragma unroll
        for (int i = 0; i < 2; ++i) {
            int r = wm * 32 + i * 16 + lrow;
            af[i].h[0] = *(const v8bf*)&lA[buf][r][lhalf * 8];
            af[i].h[1] = *(const v8bf*)&lA[buf][r][16 + lhalf * 8];
        }
#pragma unroll
        for (int j = 0; j < 4; ++j) {
            int c = wn * 64 + j * 16 + lrow;
            bfr[j].h[0] = *(const v8bf*)&lB[buf][c][lhalf * 8];
            bfr[j].h[1] = *(const v8bf*)&lB[buf][c][16 + lhalf * 8];
        }
#pragma unroll
        for (int i = 0; i < 2; ++i)
#pragma unroll
            for (int j = 0; j < 4; ++j)
                acc[i][j] = wmma_bf16(af[i].v, bfr[j].v, acc[i][j]);
        buf ^= 1;
    }

    // --- epilogue (fully resolved at compile time)
#pragma unroll
    for (int i = 0; i < 2; ++i) {
#pragma unroll
        for (int j = 0; j < 4; ++j) {
            int n = bn + wn * 64 + j * 16 + lrow;
            float bv = bias[n];
#pragma unroll
            for (int r = 0; r < 8; ++r) {
                int m = bm + wm * 32 + i * 16 + lhalf * 8 + r;
                float v = acc[i][j][r] + bv;
                if (RELU) v = fmaxf(v, 0.f);
                if (USE_RES) {
                    v += residual[(size_t)m * N + n];
                    outF[(size_t)m * N + n] = v;
                }
                if (OUTB_MODE == 1) outB[(size_t)m * N + n] = (bf16)v;
                if (OUTB_MODE == 2) outB[((size_t)(n >> 6) * M + m) * 64 + (n & 63)] = (bf16)v;
                if (OUTB_MODE == 3) outB[(size_t)n * M + m] = (bf16)v;
            }
        }
    }
}

// ---------------------------------------------------------------------------
// Flash attention: Q[H][S][64] (bf16), K[H][S][64] (bf16), Vt[H][64][S] (bf16)
// biasT[t][m] fp32 additive mask (loaded directly as the WMMA C operand).
// Out: attn bf16 [S][D] (heads re-interleaved).
// One wave per 16-query block; 32 keys per iteration; online softmax in fp32.
// P tile bounce through wave-private LDS; per-wave ordering only needs an
// in-order DS pipe + s_wait_dscnt (no workgroup barrier -> waves independent).
// ---------------------------------------------------------------------------
__global__ __launch_bounds__(256) void attn_kernel(
    const bf16* __restrict__ Q, const bf16* __restrict__ Km,
    const bf16* __restrict__ Vt, const float* __restrict__ biasT,
    bf16* __restrict__ O)
{
    __shared__ __align__(16) bf16 Plds[8][16][32];   // per-wave P tile (8KB)

    const int tid   = threadIdx.x;
    const int wave  = tid >> 5;
    const int lane  = tid & 31;
    const int lrow  = lane & 15;
    const int lhalf = lane >> 4;
    const int h     = blockIdx.y;
    const int qbase = blockIdx.x * 128 + wave * 16;

    const bf16* q  = Q  + ((size_t)h * S_LEN + qbase) * DH;
    const bf16* kp = Km + (size_t)h * S_LEN * DH;
    const bf16* vp = Vt + (size_t)h * DH * S_LEN;
    // per-lane bias pointer: row t, 8 consecutive m starting at qbase+lhalf*8
    const float* bp = biasT + qbase + lhalf * 8;

    // Q fragments: two K-chunks of 32 over Dh=64, stay register-resident
    FragAB qa[2];
#pragma unroll
    for (int kc = 0; kc < 2; ++kc) {
        const bf16* qr = q + (size_t)lrow * DH + kc * 32;
        qa[kc].h[0] = *(const v8bf*)(qr + lhalf * 8);
        qa[kc].h[1] = *(const v8bf*)(qr + 16 + lhalf * 8);
    }

    const v8f vzero = {0.f, 0.f, 0.f, 0.f, 0.f, 0.f, 0.f, 0.f};
    v8f o[4] = {vzero, vzero, vzero, vzero};
    float rm[8], rl[8];
#pragma unroll
    for (int r = 0; r < 8; ++r) { rm[r] = -3.0e38f; rl[r] = 0.f; }

    for (int kb = 0; kb < S_LEN; kb += 32) {
        // ----- scores S = Q*K^T + maskbias, two 16-key subtiles.
        // The additive mask is pre-laid-out in C-tile order: one v8f load
        // initializes the accumulator, WMMA adds QK^T on top.
        v8f sc[2];
#pragma unroll
        for (int nt = 0; nt < 2; ++nt) {
            sc[nt] = *(const v8f*)(bp + (size_t)(kb + nt * 16 + lrow) * S_LEN);
#pragma unroll
            for (int kc = 0; kc < 2; ++kc) {
                FragAB kf;
                const bf16* kr = kp + (size_t)(kb + nt * 16 + lrow) * DH + kc * 32;
                kf.h[0] = *(const v8bf*)(kr + lhalf * 8);
                kf.h[1] = *(const v8bf*)(kr + 16 + lhalf * 8);
                sc[nt] = wmma_bf16(qa[kc].v, kf.v, sc[nt]);
            }
        }
        // warm L2 for next key block
        if (kb + 32 < S_LEN) {
            __builtin_prefetch(kp + (size_t)(kb + 32 + lrow) * DH, 0, 0);
            __builtin_prefetch(vp + (size_t)lrow * S_LEN + kb + 32, 0, 0);
            __builtin_prefetch(bp + (size_t)(kb + 32 + lrow) * S_LEN, 0, 0);
        }
        // ----- online softmax: row max over 32 cols (reduce across 16-lane half)
        float tmax[8];
#pragma unroll
        for (int r = 0; r < 8; ++r) tmax[r] = fmaxf(sc[0][r], sc[1][r]);
#pragma unroll
        for (int off = 1; off < 16; off <<= 1)
#pragma unroll
            for (int r = 0; r < 8; ++r)
                tmax[r] = fmaxf(tmax[r], __shfl_xor(tmax[r], off, 32));

        float alpha[8], ts[8];
#pragma unroll
        for (int r = 0; r < 8; ++r) {
            float nm = fmaxf(rm[r], tmax[r]);
            alpha[r] = __expf(rm[r] - nm);
            rm[r] = nm;
            ts[r] = 0.f;
        }
#pragma unroll
        for (int nt = 0; nt < 2; ++nt)
#pragma unroll
            for (int r = 0; r < 8; ++r) {
                float p = __expf(sc[nt][r] - rm[r]);
                sc[nt][r] = p;
                ts[r] += p;
            }
#pragma unroll
        for (int off = 1; off < 16; off <<= 1)
#pragma unroll
            for (int r = 0; r < 8; ++r) ts[r] += __shfl_xor(ts[r], off, 32);
#pragma unroll
        for (int r = 0; r < 8; ++r) rl[r] = rl[r] * alpha[r] + ts[r];

        // ----- rescale O accumulators
#pragma unroll
        for (int j = 0; j < 4; ++j)
#pragma unroll
            for (int r = 0; r < 8; ++r) o[j][r] *= alpha[r];

        // ----- P (C-layout) -> wave-private LDS -> A-layout fragment
#pragma unroll
        for (int nt = 0; nt < 2; ++nt)
#pragma unroll
            for (int r = 0; r < 8; ++r)
                Plds[wave][lhalf * 8 + r][nt * 16 + lrow] = (bf16)sc[nt][r];
        // per-wave write->read ordering; DS pipe is in-order, no WG barrier
        asm volatile("s_wait_dscnt 0x0" ::: "memory");

        FragAB pa;
        pa.h[0] = *(const v8bf*)&Plds[wave][lrow][lhalf * 8];
        pa.h[1] = *(const v8bf*)&Plds[wave][lrow][16 + lhalf * 8];

        // ----- O += P * V  (V from transposed layout: contiguous along keys)
#pragma unroll
        for (int j = 0; j < 4; ++j) {
            FragAB vf;
            const bf16* vr = vp + (size_t)(j * 16 + lrow) * S_LEN + kb;
            vf.h[0] = *(const v8bf*)(vr + lhalf * 8);
            vf.h[1] = *(const v8bf*)(vr + 16 + lhalf * 8);
            o[j] = wmma_bf16(pa.v, vf.v, o[j]);
        }
    }

    // ----- normalize and store (re-interleave heads into [S][D])
#pragma unroll
    for (int j = 0; j < 4; ++j) {
#pragma unroll
        for (int r = 0; r < 8; ++r) {
            int m = qbase + lhalf * 8 + r;
            int d = j * 16 + lrow;
            O[(size_t)m * D_DIM + h * DH + d] = (bf16)(o[j][r] / rl[r]);
        }
    }
}

// ---------------------------------------------------------------------------
// LayerNorm over D=512: one block per row, 256 threads, 2 elems/thread.
// ---------------------------------------------------------------------------
template <bool EMIT_BF16>
__global__ __launch_bounds__(256) void layernorm_kernel(
    const float* __restrict__ x, const float* __restrict__ g,
    const float* __restrict__ b, float* __restrict__ outF,
    bf16* __restrict__ outB)
{
    __shared__ float s1[8], s2[8];
    const int row = blockIdx.x, tid = threadIdx.x;
    const float* xr = x + (size_t)row * D_DIM;
    float v0 = xr[tid], v1 = xr[tid + 256];
    float sum = v0 + v1, sq = v0 * v0 + v1 * v1;
#pragma unroll
    for (int off = 1; off < 32; off <<= 1) {
        sum += __shfl_xor(sum, off, 32);
        sq  += __shfl_xor(sq,  off, 32);
    }
    if ((tid & 31) == 0) { s1[tid >> 5] = sum; s2[tid >> 5] = sq; }
    __syncthreads();
    sum = 0.f; sq = 0.f;
#pragma unroll
    for (int i = 0; i < 8; ++i) { sum += s1[i]; sq += s2[i]; }
    float mu  = sum * (1.0f / D_DIM);
    float var = sq * (1.0f / D_DIM) - mu * mu;
    float rs  = rsqrtf(var + 1e-5f);

    float y0 = (v0 - mu) * rs * g[tid]       + b[tid];
    float y1 = (v1 - mu) * rs * g[tid + 256] + b[tid + 256];
    float* orow = outF + (size_t)row * D_DIM;
    orow[tid] = y0;
    orow[tid + 256] = y1;
    if (EMIT_BF16) {
        bf16* brow = outB + (size_t)row * D_DIM;
        brow[tid] = (bf16)y0;
        brow[tid + 256] = (bf16)y1;
    }
}

// ---------------------------------------------------------------------------
// Host launcher
// ---------------------------------------------------------------------------
extern "C" void kernel_launch(void* const* d_in, const int* in_sizes, int n_in,
                              void* d_out, int out_size, void* d_ws, size_t ws_size,
                              hipStream_t stream) {
    (void)in_sizes; (void)n_in; (void)out_size; (void)ws_size;
    const int S = S_LEN, D = D_DIM, F = F_DIM;

    const float* x    = (const float*)d_in[0];
    const int*   mask = (const int*)d_in[1];
    const float* wq = (const float*)d_in[2];  const float* bq = (const float*)d_in[3];
    const float* wk = (const float*)d_in[4];  const float* bk = (const float*)d_in[5];
    const float* wv = (const float*)d_in[6];  const float* bv = (const float*)d_in[7];
    const float* wo = (const float*)d_in[8];  const float* bo = (const float*)d_in[9];
    const float* w1 = (const float*)d_in[10]; const float* b1 = (const float*)d_in[11];
    const float* w2 = (const float*)d_in[12]; const float* b2 = (const float*)d_in[13];
    const float* g1 = (const float*)d_in[14]; const float* be1 = (const float*)d_in[15];
    const float* g2 = (const float*)d_in[16]; const float* be2 = (const float*)d_in[17];
    float* out = (float*)d_out;

    // ---- workspace bump allocator (256B aligned slices)
    char* p = (char*)d_ws;
    auto alloc = [&](size_t bytes) -> char* {
        char* r = p;
        p += (bytes + 255) & ~(size_t)255;
        return r;
    };
    bf16*  xb    = (bf16*)alloc((size_t)S * D * 2);
    bf16*  wqb   = (bf16*)alloc((size_t)D * D * 2);
    bf16*  wkb   = (bf16*)alloc((size_t)D * D * 2);
    bf16*  wvb   = (bf16*)alloc((size_t)D * D * 2);
    bf16*  wob   = (bf16*)alloc((size_t)D * D * 2);
    bf16*  w1b   = (bf16*)alloc((size_t)D * F * 2);
    bf16*  w2b   = (bf16*)alloc((size_t)F * D * 2);
    bf16*  Qh    = (bf16*)alloc((size_t)S * D * 2);   // [H][S][64]
    bf16*  Kh    = (bf16*)alloc((size_t)S * D * 2);   // [H][S][64]
    bf16*  Vt    = (bf16*)alloc((size_t)S * D * 2);   // [H][64][S]
    bf16*  attnb = (bf16*)alloc((size_t)S * D * 2);   // [S][D]
    float* t1    = (float*)alloc((size_t)S * D * 4);  // x + attn@wo + bo
    float* x1    = (float*)alloc((size_t)S * D * 4);  // LN1 output fp32
    bf16*  x1b   = (bf16*)alloc((size_t)S * D * 2);
    bf16*  hb    = (bf16*)alloc((size_t)S * F * 2);   // relu(x1@w1+b1)
    float* t2    = (float*)alloc((size_t)S * D * 4);  // x1 + ff
    float* biasT = (float*)alloc((size_t)S * S * 4);  // transposed mask bias (64MB)

    // ---- 1) convert inputs/weights to bf16; build transposed mask bias
    auto cvt = [&](const float* src, bf16* dst, int n) {
        cvt_f32_bf16<<<n / 1024, 256, 0, stream>>>(src, dst, n);
    };
    cvt(x,  xb,  S * D);
    cvt(wq, wqb, D * D); cvt(wk, wkb, D * D);
    cvt(wv, wvb, D * D); cvt(wo, wob, D * D);
    cvt(w1, w1b, D * F); cvt(w2, w2b, F * D);
    mask_bias_transpose<<<dim3(S / 32, S / 32), 256, 0, stream>>>(mask, biasT);

    // ---- 2) QKV projections (bias; head-split / transposed bf16 stores)
    dim3 blk(256);
    dim3 gDD(D / 128, S / 128);
    gemm_bf16_kernel<2, false, false><<<gDD, blk, 0, stream>>>(xb, wqb, bq, nullptr, nullptr, Qh, S, D, D);
    gemm_bf16_kernel<2, false, false><<<gDD, blk, 0, stream>>>(xb, wkb, bk, nullptr, nullptr, Kh, S, D, D);
    gemm_bf16_kernel<3, false, false><<<gDD, blk, 0, stream>>>(xb, wvb, bv, nullptr, nullptr, Vt, S, D, D);

    // ---- 3) flash attention
    attn_kernel<<<dim3(S / 128, NHEAD), blk, 0, stream>>>(Qh, Kh, Vt, biasT, attnb);

    // ---- 4) output projection + residual (fp32)
    gemm_bf16_kernel<0, false, true><<<gDD, blk, 0, stream>>>(attnb, wob, bo, x, t1, nullptr, S, D, D);

    // ---- 5) LayerNorm 1 -> x1 (fp32) + x1b (bf16)
    layernorm_kernel<true><<<S, blk, 0, stream>>>(t1, g1, be1, x1, x1b);

    // ---- 6) FFN
    dim3 gDF(F / 128, S / 128);
    gemm_bf16_kernel<1, true, false><<<gDF, blk, 0, stream>>>(x1b, w1b, b1, nullptr, nullptr, hb, S, F, D);
    gemm_bf16_kernel<0, false, true><<<gDD, blk, 0, stream>>>(hb, w2b, b2, x1, t2, nullptr, S, D, F);

    // ---- 7) LayerNorm 2 -> final output (fp32)
    layernorm_kernel<false><<<S, blk, 0, stream>>>(t2, g2, be2, out, nullptr);
}
